// CausalSelfAttention_2508260901336
// MI455X (gfx1250) — matrix operand order repaired and checked by reading, not compiled, not run on hardware
//
#include <hip/hip_runtime.h>

// CausalSelfAttention for MI455X (gfx1250, wave32, WMMA, TDM async tiles).
// B=4, S=2048, HID=1024, HEADS=16, HEAD_DIM=64.
// Pipeline: f32->bf16 cvt -> 3x WMMA GEMM (QKV, head-scattered, TDM double-buffered)
//           -> flash-attention (WMMA + online softmax, TDM K/V tiles)
//           -> WMMA GEMM (out proj, f32 to d_out).

#define BATCH 4
#define SEQ   2048
#define HID_  1024
#define HEADS 16
#define HDIM  64
#define MTOT  (BATCH * SEQ)          // 8192
#define NEGV  (-1.0e10f)

typedef __attribute__((ext_vector_type(16))) __bf16 v16bf;
typedef __attribute__((ext_vector_type(8)))  float  v8f;

union FragU { v16bf v; uint4 u[2]; };

// ---------------- TDM support (guarded; falls back to synchronous staging) ----------------
#if defined(__has_builtin)
#  if __has_builtin(__builtin_amdgcn_tensor_load_to_lds) && __has_builtin(__builtin_amdgcn_s_wait_tensorcnt)
#    define HAS_TDM 1
#  endif
#endif
#ifndef HAS_TDM
#  define HAS_TDM 0
#endif

#if HAS_TDM
typedef unsigned tdm_u32x4 __attribute__((ext_vector_type(4)));
typedef int      tdm_i32x8 __attribute__((ext_vector_type(8)));
typedef int      tdm_i32x4 __attribute__((ext_vector_type(4)));

// 2-D tile DMA: global (row-major, 2-byte elems, row stride stride_d0 elems)
//   -> LDS (packed tile_d0 x tile_d1). D# encoding per CDNA5 ISA.
// This toolchain exposes the 6-arg builtin form:
//   (u32x4 group0, i32x8 group1, i32x4, i32x4, i32x8, i32 cpol)
__device__ __forceinline__ void tdm_load_2d(const void* lds_dst, const void* gsrc,
                                            unsigned tile_d0, unsigned tile_d1,
                                            unsigned stride_d0) {
  unsigned long long ga = (unsigned long long)gsrc;
  unsigned lds = (unsigned)((unsigned long long)lds_dst & 0xFFFFFFFFull); // AS3 offset = low 32 bits
  tdm_u32x4 g0 = { 1u,                                  // count=1, user desc
                   lds,                                 // lds_addr[31:0]
                   (unsigned)(ga & 0xFFFFFFFFull),      // global_addr[31:0]
                   (unsigned)((ga >> 32) & 0x01FFFFFFull) | (2u << 30) }; // ga[56:32] | type=2
  const unsigned TD0 = 1u << 20, TD1 = 1u << 20;        // huge tensor dims -> no OOB clipping
  tdm_i32x8 g1;
  g1[0] = 0x00010000;                                   // data_size=1 (2 bytes), mask=0
  g1[1] = (int)((TD0 & 0xFFFFu) << 16);                 // tensor_dim0[15:0]
  g1[2] = (int)((TD0 >> 16) | ((TD1 & 0xFFFFu) << 16)); // tensor_dim0[31:16] | tensor_dim1[15:0]
  g1[3] = (int)((TD1 >> 16) | (tile_d0 << 16));         // tensor_dim1[31:16] | tile_dim0
  g1[4] = (int)(tile_d1 & 0xFFFFu);                     // tile_dim1, tile_dim2=0
  g1[5] = (int)stride_d0;                               // tensor_dim0_stride[31:0]
  g1[6] = 0;
  g1[7] = 0;
  tdm_i32x4 z4 = { 0, 0, 0, 0 };                        // groups 2/3 unused (2-D tensor)
  tdm_i32x8 z8 = { 0, 0, 0, 0, 0, 0, 0, 0 };
  __builtin_amdgcn_tensor_load_to_lds(g0, g1, z4, z4, z8, 0);
}
#endif

// round-to-nearest-even f32 -> bf16 bits
__device__ __forceinline__ unsigned bfbits(float x) {
  unsigned u = __float_as_uint(x);
  u += 0x7FFFu + ((u >> 16) & 1u);
  return u >> 16;
}
__device__ __forceinline__ __bf16 f2bf(float x) {
  unsigned short h = (unsigned short)bfbits(x);
  __bf16 r;
  __builtin_memcpy(&r, &h, 2);
  return r;
}

// ---------------- conversion kernel: float4 -> 4x bf16 packed ----------------
__global__ void cvt_f32_bf16(const float4* __restrict__ in, uint2* __restrict__ out, int n4) {
  int i = blockIdx.x * blockDim.x + threadIdx.x;
  int stride = gridDim.x * blockDim.x;
  for (; i < n4; i += stride) {
    float4 f = in[i];
    uint2 p;
    p.x = bfbits(f.x) | (bfbits(f.y) << 16);
    p.y = bfbits(f.z) | (bfbits(f.w) << 16);
    out[i] = p;
  }
}

// ---------------- fragment loaders (layouts per ISA 7.12.2) ----------------
// A-matrix 16x32 bf16: lane m (0-15) holds K {0..7, 16..23}; lane m+16 holds K {8..15, 24..31}
__device__ __forceinline__ v16bf load_afrag(const __bf16* base, int stride_elems, int lane) {
  int m  = lane & 15;
  int k0 = (lane < 16) ? 0 : 8;
  FragU f;
  f.u[0] = *reinterpret_cast<const uint4*>(base + m * stride_elems + k0);
  f.u[1] = *reinterpret_cast<const uint4*>(base + m * stride_elems + k0 + 16);
  return f.v;
}
// B-matrix 32x16 bf16: lane n (0-15) holds col n, K {0..7, 16..23}; lane n+16 holds K {8..15, 24..31}
__device__ __forceinline__ v16bf load_bfrag(const __bf16* base, int stride_elems, int lane) {
  int n  = lane & 15;
  int k0 = (lane < 16) ? 0 : 8;
  FragU f;
  f.u[0] = *reinterpret_cast<const uint4*>(base + n * stride_elems + k0);
  f.u[1] = *reinterpret_cast<const uint4*>(base + n * stride_elems + k0 + 16);
  return f.v;
}

// ---------------- WMMA GEMM:  C[M,N] = A[M,K] @ W[N,K]^T + bias ----------------
// Tile: BM=128, BN=128, BK=32; 8 waves, each computing 32 rows x 64 cols (2x4 frags).
// MODE 1: bf16 output scattered to [B, HEADS, SEQ, HDIM]   (QKV projections)
// MODE 2: f32 output, plain row-major [M, N]               (final projection)
template <int MODE>
__global__ __launch_bounds__(256)
void gemm_bf16_wmma(const __bf16* __restrict__ A, const __bf16* __restrict__ W,
                    const float* __restrict__ bias, void* __restrict__ outp) {
  constexpr int K  = HID_;
  constexpr int KT = K / 32;
  __shared__ __align__(16) __bf16 As[2][128 * 32];
  __shared__ __align__(16) __bf16 Bs[2][128 * 32];

  const int tid  = threadIdx.x;
  const int lane = tid & 31;
  const int wave = tid >> 5;       // 0..7
  const int wm   = wave & 3;       // row quarter (32 rows)
  const int wn   = wave >> 2;      // col half   (64 cols)
  const int m0   = blockIdx.y * 128;
  const int n0   = blockIdx.x * 128;

  v8f c[2][4] = {};

#if HAS_TDM
  if (wave == 0) {
    tdm_load_2d(&As[0][0], A + (size_t)m0 * K, 32, 128, K);
    tdm_load_2d(&Bs[0][0], W + (size_t)n0 * K, 32, 128, K);
  }
#endif

  for (int kt = 0; kt < KT; ++kt) {
#if HAS_TDM
    const int buf = kt & 1;
    __syncthreads();   // everyone done reading buf^1 from iteration kt-1
    if (wave == 0) {
      if (kt + 1 < KT) {
        tdm_load_2d(&As[buf ^ 1][0], A + (size_t)m0 * K + (kt + 1) * 32, 32, 128, K);
        tdm_load_2d(&Bs[buf ^ 1][0], W + (size_t)n0 * K + (kt + 1) * 32, 32, 128, K);
        __builtin_amdgcn_s_wait_tensorcnt(2);  // in-order: pair for tile kt has landed
      } else {
        __builtin_amdgcn_s_wait_tensorcnt(0);
      }
    }
    __syncthreads();   // tile kt visible to all waves
#else
    const int buf = 0;
    __syncthreads();
    {
      int row  = tid >> 1;                 // 0..127
      int half = (tid & 1) * 16;
      const uint4* asrc = reinterpret_cast<const uint4*>(A + (size_t)(m0 + row) * K + kt * 32 + half);
      uint4 a0 = asrc[0], a1 = asrc[1];
      uint4* adst = reinterpret_cast<uint4*>(&As[0][row * 32 + half]);
      adst[0] = a0; adst[1] = a1;
      const uint4* bsrc = reinterpret_cast<const uint4*>(W + (size_t)(n0 + row) * K + kt * 32 + half);
      uint4 b0 = bsrc[0], b1 = bsrc[1];
      uint4* bdst = reinterpret_cast<uint4*>(&Bs[0][row * 32 + half]);
      bdst[0] = b0; bdst[1] = b1;
    }
    __syncthreads();
#endif

    v16bf bfr[4];
#pragma unroll
    for (int j = 0; j < 4; ++j)
      bfr[j] = load_bfrag(&Bs[buf][(wn * 64 + j * 16) * 32], 32, lane);
#pragma unroll
    for (int i = 0; i < 2; ++i) {
      v16bf af = load_afrag(&As[buf][(wm * 32 + i * 16) * 32], 32, lane);
#pragma unroll
      for (int j = 0; j < 4; ++j)
        c[i][j] = __builtin_amdgcn_wmma_f32_16x16x32_bf16(
            false, af, false, bfr[j], (short)0, c[i][j], false, false);
    }
  }

  // epilogue: C frag layout -> lane holds col n=lane%16; VGPR r -> row r (+8 for upper half)
  const int nlane = lane & 15;
  const int rbase = (lane < 16) ? 0 : 8;
#pragma unroll
  for (int i = 0; i < 2; ++i) {
#pragma unroll
    for (int j = 0; j < 4; ++j) {
      int ng = n0 + wn * 64 + j * 16 + nlane;
      float bv = bias[ng];
#pragma unroll
      for (int r = 0; r < 8; ++r) {
        int mg = m0 + wm * 32 + i * 16 + rbase + r;
        float val = c[i][j][r] + bv;
        if (MODE == 1) {
          int b = mg >> 11, s = mg & (SEQ - 1);
          int h = ng >> 6,  d = ng & (HDIM - 1);
          ((__bf16*)outp)[(((size_t)(b * HEADS + h) * SEQ) + s) * HDIM + d] = f2bf(val);
        } else {
          ((float*)outp)[(size_t)mg * HID_ + ng] = val;
        }
      }
    }
  }
}

// ---------------- Flash attention (causal, unscaled logits like the reference) ----------------
// grid: (SEQ/64, BATCH*HEADS), block: 128 (4 waves). Each wave owns 16 q-rows.
// Q/K/V in [B, HEADS, SEQ, HDIM] bf16; O written as [B, SEQ, HEADS*HDIM] bf16.
__global__ __launch_bounds__(128)
void flash_attn_wmma(const __bf16* __restrict__ Q, const __bf16* __restrict__ K,
                     const __bf16* __restrict__ V, __bf16* __restrict__ O) {
  __shared__ __align__(16) __bf16 Ks[2][32 * 64];     // [kv][d]
  __shared__ __align__(16) __bf16 Vs[2][32 * 64];     // [kv][d]  (TDM landing zone)
  __shared__ __align__(16) __bf16 Vts[2][64 * 32];    // [d][kv]  (transposed for B-frag reads)
  __shared__ __align__(16) __bf16 Ps[4][16 * 32];     // per-wave P tile [m][kv]

  const int tid  = threadIdx.x;
  const int lane = tid & 31;
  const int wave = tid >> 5;     // 0..3
  const int bh   = blockIdx.y;   // b*HEADS + h
  const int q0   = blockIdx.x * 64;

  const __bf16* Qb = Q + (size_t)bh * SEQ * HDIM;
  const __bf16* Kb = K + (size_t)bh * SEQ * HDIM;
  const __bf16* Vb = V + (size_t)bh * SEQ * HDIM;

  const int nlane = lane & 15;
  const int rbase = (lane < 16) ? 0 : 8;
  const int qrow0 = q0 + wave * 16;      // this wave's 16 q-rows

  // Q A-fragments, two K-chunks covering D=64
  v16bf qf[2];
  {
    const __bf16* qr = Qb + (size_t)(qrow0 + nlane) * HDIM;
    int kh = (lane < 16) ? 0 : 8;
    FragU f0, f1;
    f0.u[0] = *reinterpret_cast<const uint4*>(qr + kh);
    f0.u[1] = *reinterpret_cast<const uint4*>(qr + kh + 16);
    f1.u[0] = *reinterpret_cast<const uint4*>(qr + 32 + kh);
    f1.u[1] = *reinterpret_cast<const uint4*>(qr + 32 + kh + 16);
    qf[0] = f0.v; qf[1] = f1.v;
  }

  float mrow[8], lrow[8];
  v8f acc[4] = {};                 // 4 d-tiles of 16
#pragma unroll
  for (int r = 0; r < 8; ++r) { mrow[r] = -1.0e30f; lrow[r] = 0.0f; }

  const int nkv = (q0 + 64) / 32;  // causal: kv < q0+64 (always >= 2)

#if HAS_TDM
  if (wave == 0) {
    tdm_load_2d(&Ks[0][0], Kb, HDIM, 32, HDIM);
    tdm_load_2d(&Vs[0][0], Vb, HDIM, 32, HDIM);
  }
#endif

  for (int kb = 0; kb < nkv; ++kb) {
    const int kv0 = kb * 32;
#if HAS_TDM
    const int buf = kb & 1;
    __syncthreads();   // all waves done with buf^1 tiles from iteration kb-1
    if (wave == 0) {
      if (kb + 1 < nkv) {
        tdm_load_2d(&Ks[buf ^ 1][0], Kb + (size_t)(kv0 + 32) * HDIM, HDIM, 32, HDIM);
        tdm_load_2d(&Vs[buf ^ 1][0], Vb + (size_t)(kv0 + 32) * HDIM, HDIM, 32, HDIM);
        __builtin_amdgcn_s_wait_tensorcnt(2);  // tile kb has landed
      } else {
        __builtin_amdgcn_s_wait_tensorcnt(0);
      }
    }
    __syncthreads();   // Ks[buf]/Vs[buf] visible
    // transpose V tile LDS->LDS: [kv][d] -> [d][kv]
    {
      int row = tid >> 2;               // 0..31
      int ch  = (tid & 3) * 16;         // 0,16,32,48
      FragU f;
      const __bf16* vsrc = &Vs[buf][row * 64 + ch];
      f.u[0] = *reinterpret_cast<const uint4*>(vsrc);
      f.u[1] = *reinterpret_cast<const uint4*>(vsrc + 8);
#pragma unroll
      for (int e = 0; e < 16; ++e)
        Vts[buf][(ch + e) * 32 + row] = f.v[e];
    }
#else
    const int buf = 0;
    __syncthreads();
    {
      int row = tid >> 2;               // 0..31
      int ch  = (tid & 3) * 16;         // 0,16,32,48
      const uint4* ksrc = reinterpret_cast<const uint4*>(Kb + (size_t)(kv0 + row) * HDIM + ch);
      uint4 k0v = ksrc[0], k1v = ksrc[1];
      uint4* kdst = reinterpret_cast<uint4*>(&Ks[0][row * 64 + ch]);
      kdst[0] = k0v; kdst[1] = k1v;
      FragU f;
      const __bf16* vsrc = Vb + (size_t)(kv0 + row) * HDIM + ch;
      f.u[0] = *reinterpret_cast<const uint4*>(vsrc);
      f.u[1] = *reinterpret_cast<const uint4*>(vsrc + 8);
#pragma unroll
      for (int e = 0; e < 16; ++e)
        Vts[0][(ch + e) * 32 + row] = f.v[e];
    }
#endif
    __syncthreads();   // K tile + V^T tile ready for all waves

    // S = Q . K^T : two 16-col score tiles, D=64 -> 2 chained WMMAs each
    v8f sc[2];
#pragma unroll
    for (int t = 0; t < 2; ++t) {
      v8f s = {};
#pragma unroll
      for (int kc = 0; kc < 2; ++kc) {
        int kk = kc * 32 + ((lane < 16) ? 0 : 8);
        FragU f;
        const __bf16* base = &Ks[buf][(t * 16 + nlane) * 64 + kk];
        f.u[0] = *reinterpret_cast<const uint4*>(base);
        f.u[1] = *reinterpret_cast<const uint4*>(base + 16);
        s = __builtin_amdgcn_wmma_f32_16x16x32_bf16(
            false, qf[kc], false, f.v, (short)0, s, false, false);
      }
      sc[t] = s;
    }

    // causal mask (only near the diagonal)
    if (kv0 + 31 > qrow0) {
#pragma unroll
      for (int t = 0; t < 2; ++t) {
        int kvg = kv0 + t * 16 + nlane;
#pragma unroll
        for (int r = 0; r < 8; ++r) {
          int qg = qrow0 + rbase + r;
          if (kvg > qg) sc[t][r] = NEGV;
        }
      }
    }

    // online softmax: row reductions across the 16-lane half (xor 1,2,4,8)
    float newm[8], alpha[8];
#pragma unroll
    for (int r = 0; r < 8; ++r) {
      float mx = fmaxf(sc[0][r], sc[1][r]);
#pragma unroll
      for (int off = 8; off >= 1; off >>= 1)
        mx = fmaxf(mx, __shfl_xor(mx, off, 32));
      newm[r]  = fmaxf(mrow[r], mx);
      alpha[r] = __expf(mrow[r] - newm[r]);
    }
#pragma unroll
    for (int r = 0; r < 8; ++r) {
      float p0 = __expf(sc[0][r] - newm[r]);
      float p1 = __expf(sc[1][r] - newm[r]);
      sc[0][r] = p0; sc[1][r] = p1;
      float sum = p0 + p1;
#pragma unroll
      for (int off = 8; off >= 1; off >>= 1)
        sum += __shfl_xor(sum, off, 32);
      lrow[r] = lrow[r] * alpha[r] + sum;
      mrow[r] = newm[r];
    }
#pragma unroll
    for (int dt = 0; dt < 4; ++dt)
#pragma unroll
      for (int r = 0; r < 8; ++r)
        acc[dt][r] *= alpha[r];

    // write P (bf16) to wave-local LDS, re-read in A-fragment layout
    __bf16* pw = &Ps[wave][0];
#pragma unroll
    for (int t = 0; t < 2; ++t)
#pragma unroll
      for (int r = 0; r < 8; ++r)
        pw[(rbase + r) * 32 + t * 16 + nlane] = f2bf(sc[t][r]);
    asm volatile("" ::: "memory");  // keep DS store->load program order (same-wave, in-order pipe)
    v16bf pf;
    {
      int kh = (lane < 16) ? 0 : 8;
      FragU f;
      const __bf16* base = pw + nlane * 32 + kh;
      f.u[0] = *reinterpret_cast<const uint4*>(base);
      f.u[1] = *reinterpret_cast<const uint4*>(base + 16);
      pf = f.v;
    }

    // O += P . V  (K = 32 -> one WMMA per 16-wide d-tile)
#pragma unroll
    for (int dt = 0; dt < 4; ++dt) {
      int kh = (lane < 16) ? 0 : 8;
      FragU f;
      const __bf16* base = &Vts[buf][(dt * 16 + nlane) * 32 + kh];
      f.u[0] = *reinterpret_cast<const uint4*>(base);
      f.u[1] = *reinterpret_cast<const uint4*>(base + 16);
      acc[dt] = __builtin_amdgcn_wmma_f32_16x16x32_bf16(
          false, pf, false, f.v, (short)0, acc[dt], false, false);
    }
  }

  // normalize & store O as [B, SEQ, HEADS*HDIM]
  const int b = bh >> 4;
  const int h = bh & 15;
#pragma unroll
  for (int dt = 0; dt < 4; ++dt) {
    int d = dt * 16 + nlane;
#pragma unroll
    for (int r = 0; r < 8; ++r) {
      int qg = qrow0 + rbase + r;
      float o = acc[dt][r] / lrow[r];
      O[((size_t)(b * SEQ + qg)) * HID_ + h * HDIM + d] = f2bf(o);
    }
  }
}

// ---------------- launch ----------------
extern "C" void kernel_launch(void* const* d_in, const int* in_sizes, int n_in,
                              void* d_out, int out_size, void* d_ws, size_t ws_size,
                              hipStream_t stream) {
  (void)in_sizes; (void)n_in; (void)out_size; (void)ws_size;
  const float* q  = (const float*)d_in[0];
  const float* k  = (const float*)d_in[1];
  const float* v  = (const float*)d_in[2];
  const float* Wq = (const float*)d_in[3];
  const float* bq = (const float*)d_in[4];
  const float* Wk = (const float*)d_in[5];
  const float* bk = (const float*)d_in[6];
  const float* Wv = (const float*)d_in[7];
  const float* bv = (const float*)d_in[8];
  const float* Wo = (const float*)d_in[9];
  const float* bo = (const float*)d_in[10];
  float* out = (float*)d_out;

  __bf16* ws = (__bf16*)d_ws;
  const size_t WE = (size_t)HID_ * HID_;   // 1,048,576
  const size_t XE = (size_t)MTOT * HID_;   // 8,388,608
  __bf16* Wq_b = ws;
  __bf16* Wk_b = Wq_b + WE;
  __bf16* Wv_b = Wk_b + WE;
  __bf16* Wo_b = Wv_b + WE;
  __bf16* Xq   = Wo_b + WE;
  __bf16* Xk   = Xq + XE;
  __bf16* Xv   = Xk + XE;
  __bf16* Qh   = Xv + XE;
  __bf16* Kh   = Qh + XE;
  __bf16* Vh   = Kh + XE;
  __bf16* Oh   = Vh + XE;

  // f32 -> bf16 conversions (vectorized: float4 -> packed 2x u32)
  cvt_f32_bf16<<<2048, 256, 0, stream>>>((const float4*)q,  (uint2*)Xq,   (int)(XE / 4));
  cvt_f32_bf16<<<2048, 256, 0, stream>>>((const float4*)k,  (uint2*)Xk,   (int)(XE / 4));
  cvt_f32_bf16<<<2048, 256, 0, stream>>>((const float4*)v,  (uint2*)Xv,   (int)(XE / 4));
  cvt_f32_bf16<<<512,  256, 0, stream>>>((const float4*)Wq, (uint2*)Wq_b, (int)(WE / 4));
  cvt_f32_bf16<<<512,  256, 0, stream>>>((const float4*)Wk, (uint2*)Wk_b, (int)(WE / 4));
  cvt_f32_bf16<<<512,  256, 0, stream>>>((const float4*)Wv, (uint2*)Wv_b, (int)(WE / 4));
  cvt_f32_bf16<<<512,  256, 0, stream>>>((const float4*)Wo, (uint2*)Wo_b, (int)(WE / 4));

  // QKV projections (head-scattered outputs)
  dim3 ggrid(HID_ / 128, MTOT / 128);   // (8, 64)
  gemm_bf16_wmma<1><<<ggrid, 256, 0, stream>>>(Xq, Wq_b, bq, (void*)Qh);
  gemm_bf16_wmma<1><<<ggrid, 256, 0, stream>>>(Xk, Wk_b, bk, (void*)Kh);
  gemm_bf16_wmma<1><<<ggrid, 256, 0, stream>>>(Xv, Wv_b, bv, (void*)Vh);

  // flash attention
  flash_attn_wmma<<<dim3(SEQ / 64, BATCH * HEADS), 128, 0, stream>>>(Qh, Kh, Vh, Oh);

  // output projection -> f32 d_out
  gemm_bf16_wmma<2><<<ggrid, 256, 0, stream>>>(Oh, Wo_b, bo, (void*)out);
}
